// MADGraph_4612794876396
// MI455X (gfx1250) — matrix operand order repaired and checked by reading, not compiled, not run on hardware
//
#include <hip/hip_runtime.h>
#include <math.h>

typedef __attribute__((ext_vector_type(16))) _Float16 v16h;
typedef __attribute__((ext_vector_type(8)))  float    v8f;

#define N_NODES   10000
#define FEATS     32
#define HEADS     4
#define SAMPLES   256
#define SENTINELS 8
#define N_BATCH   2048

// Zero the output buffer (graph-safe, deterministic).
__global__ void madg_zero_kernel(float* out, int n) {
    int i = blockIdx.x * blockDim.x + threadIdx.x;
    if (i < n) out[i] = 0.0f;
}

// One block per (n, h). 256 threads = 8 wave32s. Each wave does 4 sample-tiles
// of 16 samples; each tile is one v_wmma_f32_16x16x32_f16 computing, per
// sample row, dot(pos[mid], field_row) and dot(pos[mid], pos_row).
__global__ void __launch_bounds__(256)
madg_main_kernel(const int* __restrict__ edge,
                 const int* __restrict__ mid0,
                 const int* __restrict__ mid1,
                 const float* __restrict__ pos,
                 const float* __restrict__ field,
                 const float* __restrict__ unc,
                 const float* __restrict__ edge_mat,
                 float* __restrict__ out) {
    const int n = blockIdx.x;   // edge index
    const int h = blockIdx.y;   // head

    __shared__ float sPsrc[FEATS], sPdst[FEATS], sFsrc[FEATS], sFdst[FEATS];
    __shared__ float sC[4];            // p_src.f_dst, p_dst.f_src, |p_src|^2, |p_dst|^2
    __shared__ float sUnc;
    __shared__ float sCt[8][16][16];   // per-wave full C tile [row][col]
    __shared__ float sScore[2 * SAMPLES];
    __shared__ float sLogit[2 * SAMPLES];
    __shared__ float sRedA[256], sRedB[256];

    const int tid  = threadIdx.x;
    const int wave = tid >> 5;
    const int lane = tid & 31;

    const int esrc = edge[2 * n + 0];
    const int edst = edge[2 * n + 1];

    // ---- Prologue: stage the four per-(h,n) rows in LDS ----
    if (tid < FEATS) {
        const size_t bs = ((size_t)h * N_NODES + esrc) * FEATS + tid;
        const size_t bd = ((size_t)h * N_NODES + edst) * FEATS + tid;
        sPsrc[tid] = pos[bs];
        sPdst[tid] = pos[bd];
        sFsrc[tid] = field[bs];
        sFdst[tid] = field[bd];
    }
    if (tid == 0) sUnc = unc[0];
    __syncthreads();

    if (tid < 32) {
        float ps = sPsrc[tid], pd = sPdst[tid], fs = sFsrc[tid], fd = sFdst[tid];
        float v1 = ps * fd, v2 = pd * fs, v3 = ps * ps, v4 = pd * pd;
        #pragma unroll
        for (int off = 16; off >= 1; off >>= 1) {
            v1 += __shfl_down(v1, off);
            v2 += __shfl_down(v2, off);
            v3 += __shfl_down(v3, off);
            v4 += __shfl_down(v4, off);
        }
        if (tid == 0) { sC[0] = v1; sC[1] = v2; sC[2] = v3; sC[3] = v4; }
    }
    __syncthreads();

    const int msel = lane & 15;   // A: row-in-tile; B: column index; C: column index
    const int hi   = lane >> 4;   // lane half (K-split select)

    // ---- Build both B matrices ONCE (side 0: [f_dst, p_src], side 1: [f_src, p_dst]) ----
    // B layout: lane holds column msel; lanes<16 K={0..7,16..23}, lanes>=16 K={8..15,24..31}.
    v16h bv[2];
    {
        const int kb = hi * 8;
        #pragma unroll
        for (int sd = 0; sd < 2; ++sd) {
            const float* col0 = sd ? sFsrc : sFdst;
            const float* col1 = sd ? sPdst : sPsrc;
            v16h b;
            #pragma unroll
            for (int j = 0; j < 16; ++j) {
                float x = (msel == 0) ? col0[kb + j + (j >= 8 ? 8 : 0)]
                        : (msel == 1) ? col1[kb + j + (j >= 8 ? 8 : 0)] : 0.0f;
                b[j] = (_Float16)x;
            }
            bv[sd] = b;
        }
    }

    const size_t midBase = ((size_t)h * N_BATCH + n) * SAMPLES;
    const int klo = hi * 8;

    // ---- Tile loop: side is a compile-time constant in the unrolled body ----
    #pragma unroll
    for (int side = 0; side < 2; ++side) {
        const int* midp = side ? mid1 : mid0;
        #pragma unroll
        for (int i = 0; i < 2; ++i) {
            const int tt = wave * 2 + i;      // tile within side: 0..15
            const int t  = side * 16 + tt;    // global tile: 0..31
            const int s  = tt * 16 + msel;    // sample index
            const int m  = midp[midBase + s];

            // Prefetch the edge_mat scalar for this sample (HBM-latency hiding).
            const float* emp = side ? (edge_mat + (size_t)esrc * N_NODES + m)
                                    : (edge_mat + (size_t)m * N_NODES + edst);
            __builtin_prefetch(emp, 0, 0);

            // ---- A matrix: gathered pos row (f32 -> f16), ISA 16-bit A layout ----
            // lanes<16: K in {0..7, 16..23}; lanes>=16: K in {8..15, 24..31}
            const float* prow = pos + ((size_t)h * N_NODES + m) * FEATS;
            float av[16];
            #pragma unroll
            for (int j = 0; j < 8; ++j) av[j] = prow[klo + j];
            #pragma unroll
            for (int j = 0; j < 8; ++j) av[8 + j] = prow[16 + klo + j];

            float ss = 0.0f;
            #pragma unroll
            for (int j = 0; j < 16; ++j) ss += av[j] * av[j];
            const float nrm2 = ss + __shfl_xor(ss, 16);  // full |pos[m]|^2

            v16h a;
            #pragma unroll
            for (int j = 0; j < 16; ++j) a[j] = (_Float16)av[j];

            // ---- 16x16x32 matrix op: C[row][0]=pos_m.field, C[row][1]=pos_m.pos ----
            v8f c = {};
            c = __builtin_amdgcn_wmma_f32_16x16x32_f16(
                /*neg_a=*/false, a, /*neg_b=*/false, bv[side],
                /*c_mod=*/(short)0, c, /*reuse_a=*/false, /*reuse_b=*/false);

            // ---- Branch-free C extraction: store the full tile ----
            // sCt is per-wave; same-wave LDS ops execute in order (DScnt), so
            // no block barrier is needed around these accesses.
            #pragma unroll
            for (int r = 0; r < 8; ++r) sCt[wave][r + 8 * hi][msel] = c[r];

            // ---- Per-sample finalize (lanes 0..15 own one sample each) ----
            if (lane < 16) {
                const float dF = sCt[wave][lane][0];
                const float dP = sCt[wave][lane][1];
                const float em = *emp;
                const float cc = side ? sC[1] : sC[0];
                const float nn = side ? sC[3] : sC[2];
                const float logit = cc - dF + sUnc * em;
                float d2 = nn - 2.0f * dP + nrm2;
                d2 = fmaxf(d2, 0.0f);
                const int e = t * 16 + lane;
                sScore[e] = 1.0f - sqrtf(d2);
                sLogit[e] = logit;
            }
        }
    }
    __syncthreads();

    // ---- Softmax over 512 computed entries + 8 sentinels (score 0, logit 0) ----
    float lm = 0.0f;  // sentinel score = 1 - 1 = 0 participates in the max
    for (int e = tid; e < 2 * SAMPLES; e += 256) lm = fmaxf(lm, sScore[e]);
    sRedA[tid] = lm;
    __syncthreads();
    #pragma unroll
    for (int s2 = 128; s2 > 0; s2 >>= 1) {
        if (tid < s2) sRedA[tid] = fmaxf(sRedA[tid], sRedA[tid + s2]);
        __syncthreads();
    }
    const float mx = sRedA[0];
    __syncthreads();

    float zs = 0.0f, ns = 0.0f;
    for (int e = tid; e < 2 * SAMPLES; e += 256) {
        const float w = __expf(sScore[e] - mx);
        zs += w;
        ns += w * sLogit[e];
    }
    sRedA[tid] = zs;
    sRedB[tid] = ns;
    __syncthreads();
    #pragma unroll
    for (int s2 = 128; s2 > 0; s2 >>= 1) {
        if (tid < s2) {
            sRedA[tid] += sRedA[tid + s2];
            sRedB[tid] += sRedB[tid + s2];
        }
        __syncthreads();
    }

    if (tid == 0) {
        const float Z = sRedA[0] + (float)SENTINELS * __expf(-mx);
        atomicAdd(&out[n], 0.25f * (sRedB[0] / Z));  // mean over HEADS=4
    }
}

extern "C" void kernel_launch(void* const* d_in, const int* in_sizes, int n_in,
                              void* d_out, int out_size, void* d_ws, size_t ws_size,
                              hipStream_t stream) {
    (void)in_sizes; (void)n_in; (void)d_ws; (void)ws_size;
    const int*   edge     = (const int*)d_in[0];
    const int*   mid0     = (const int*)d_in[1];
    const int*   mid1     = (const int*)d_in[2];
    const float* pos      = (const float*)d_in[3];
    const float* field    = (const float*)d_in[4];
    const float* unc      = (const float*)d_in[5];
    const float* edge_mat = (const float*)d_in[6];
    float*       out      = (float*)d_out;

    madg_zero_kernel<<<(out_size + 255) / 256, 256, 0, stream>>>(out, out_size);

    dim3 grid(N_BATCH, HEADS);
    madg_main_kernel<<<grid, 256, 0, stream>>>(edge, mid0, mid1, pos, field, unc,
                                               edge_mat, out);
}
